// CrossAttention_23021024706948
// MI455X (gfx1250) — matrix operand order repaired and checked by reading, not compiled, not run on hardware
//
#include <hip/hip_runtime.h>
#include <hip/hip_bf16.h>

// ---------------------------------------------------------------------------
// CDNA5 (gfx1250) cross-attention with Euclidean-distance similarity.
// f16 WMMA (v_wmma_f32_16x16x32_f16) for all GEMMs + flash-style attention.
// Double-buffered async global->LDS staging (ASYNCcnt) for attention K tiles.
// ---------------------------------------------------------------------------

typedef __attribute__((ext_vector_type(16))) _Float16 v16h;
typedef __attribute__((ext_vector_type(8)))  _Float16 v8h;
typedef __attribute__((ext_vector_type(8)))  float    v8f;

union F16x16 { v16h v; v8h h[2]; };

// A-matrix fragment, 16x32 f16 (ISA 7.12.2):
// lane L: m = L&15 ; halves e=0..7 -> k = koff+e ; e=8..15 -> k = koff+16+(e-8)
// where koff = (L&16)?8:0.  Source is row-major [m][k] with row stride `ld`.
static __device__ inline v16h load_a_frag(const _Float16* base, int ld, int lane) {
  const _Float16* p = base + (size_t)(lane & 15) * ld + ((lane & 16) ? 8 : 0);
  F16x16 r;
  r.h[0] = *(const v8h*)(p);
  r.h[1] = *(const v8h*)(p + 16);
  return r.v;
}

// B-matrix fragment, 32x16 f16: lane L: n = L&15 ; e=0..7 -> k = kb+e ; e=8..15 -> k = kb+16+(e-8),
// kb = (L&16)?8:0.  Source stored as [n][k] row-major ("W[N][K]"), stride ld.
static __device__ inline v16h load_b_frag(const _Float16* base, int ld, int lane) {
  const _Float16* p = base + (size_t)(lane & 15) * ld + ((lane & 16) ? 8 : 0);
  F16x16 r;
  r.h[0] = *(const v8h*)(p);
  r.h[1] = *(const v8h*)(p + 16);
  return r.v;
}

static __device__ inline v8f wmma_f16(v16h a, v16h b, v8f c) {
  // 8 args: (neg_a, A, neg_b, B, c_mod, C, reuse_a, reuse_b)
  return __builtin_amdgcn_wmma_f32_16x16x32_f16(false, a, false, b, (short)0, c,
                                                false, false);
}

// ---------------------------------------------------------------------------
// fp32 -> f16 conversion (weights)
// ---------------------------------------------------------------------------
__global__ __launch_bounds__(256) void f32_to_f16_kernel(const float* __restrict__ x,
                                                         _Float16* __restrict__ y,
                                                         int n) {
  int i = blockIdx.x * 256 + threadIdx.x;
  if (i < n) y[i] = (_Float16)x[i];
}

// ---------------------------------------------------------------------------
// Channel RMS-norm of fmap [b][c=512][xy=1024]  ->  fm_n f16 [b][xy][c]
// ---------------------------------------------------------------------------
__global__ __launch_bounds__(256) void norm_fmap_kernel(const float* __restrict__ fmap,
                                                        const float* __restrict__ gamma,
                                                        _Float16* __restrict__ out) {
  const int xy = blockIdx.x * 256 + threadIdx.x;  // grid.x = 4
  const int b  = blockIdx.y;                      // grid.y = 8
  const float* base = fmap + (size_t)b * 512 * 1024;
  float s = 0.f;
  for (int c = 0; c < 512; ++c) {
    float v = base[(size_t)c * 1024 + xy];
    s += v * v;
  }
  const float scale = 22.62741699796952f / fmaxf(sqrtf(s), 1e-12f);  // sqrt(512)
  _Float16* o = out + ((size_t)b * 1024 + xy) * 512;
  for (int c0 = 0; c0 < 512; c0 += 8) {
    v8h pack;
    for (int u = 0; u < 8; ++u)
      pack[u] = (_Float16)(base[(size_t)(c0 + u) * 1024 + xy] * scale * gamma[c0 + u]);
    *(v8h*)(o + c0) = pack;
  }
}

// ---------------------------------------------------------------------------
// Row RMS-norm of context [row][512] -> ctx_n f16 [row][512]
// ---------------------------------------------------------------------------
__global__ __launch_bounds__(256) void norm_rows_kernel(const float* __restrict__ x,
                                                        const float* __restrict__ gamma,
                                                        _Float16* __restrict__ y) {
  const int row = blockIdx.x;
  const float* xr = x + (size_t)row * 512;
  float s = 0.f;
  for (int c = threadIdx.x; c < 512; c += 256) {
    float v = xr[c];
    s += v * v;
  }
  __shared__ float red[256];
  red[threadIdx.x] = s;
  __syncthreads();
  for (int off = 128; off > 0; off >>= 1) {
    if (threadIdx.x < off) red[threadIdx.x] += red[threadIdx.x + off];
    __syncthreads();
  }
  const float scale = 22.62741699796952f / fmaxf(sqrtf(red[0]), 1e-12f);
  _Float16* yr = y + (size_t)row * 512;
  for (int c = threadIdx.x; c < 512; c += 256)
    yr[c] = (_Float16)(xr[c] * scale * gamma[c]);
}

// ---------------------------------------------------------------------------
// Row-of-64 squared-norm: out[(b*8+h)*1024 + i] = sum_d data[b][i][colbase+h*64+d]^2
// ---------------------------------------------------------------------------
__global__ __launch_bounds__(256) void rowsq_kernel(const _Float16* __restrict__ data,
                                                    float* __restrict__ out,
                                                    int ld, int colbase) {
  const int idx = blockIdx.x * 256 + threadIdx.x;  // 65536 total
  const int i = idx & 1023;
  const int h = (idx >> 10) & 7;
  const int b = idx >> 13;
  const _Float16* p = data + ((size_t)(b * 1024 + i)) * ld + colbase + h * 64;
  float s = 0.f;
  for (int d0 = 0; d0 < 64; d0 += 8) {
    v8h pk = *(const v8h*)(p + d0);
    for (int u = 0; u < 8; ++u) {
      float v = (float)pk[u];
      s += v * v;
    }
  }
  out[((size_t)(b * 8 + h)) * 1024 + i] = s;
}

// ---------------------------------------------------------------------------
// GEMM: C[b][M][N] = A[b][M][K] (f16, lda=K) x W[N][K]^T (f16, contiguous K).
// Block = 8 waves; wave w -> 16x64 strip at (blockIdx.y*128 + w*16, blockIdx.x*64).
// mode 0: store f16 row-major [M][N].  mode 1: store f32 transposed [N][M].
// ---------------------------------------------------------------------------
__global__ __launch_bounds__(256) void gemm_f16_kernel(
    const _Float16* __restrict__ A, const _Float16* __restrict__ W,
    void* __restrict__ Cout, int M, int N, int K,
    size_t strideA, size_t strideC, int mode) {
  const int lane = threadIdx.x & 31;
  const int wave = threadIdx.x >> 5;
  const int m0 = blockIdx.y * 128 + wave * 16;
  const int n0 = blockIdx.x * 64;
  const int b  = blockIdx.z;
  const _Float16* Ab = A + (size_t)b * strideA + (size_t)m0 * K;
  const _Float16* Apf = Ab + (size_t)(lane & 15) * K;  // per-lane prefetch base

  v8f acc0 = {}, acc1 = {}, acc2 = {}, acc3 = {};
  for (int k0 = 0; k0 < K; k0 += 32) {
    // WGP-scope speculative prefetch of the A fragment two k-steps ahead
    // (locality 3 -> fill all cache levels; OOB speculative prefetch dropped).
    __builtin_prefetch(Apf + k0 + 64, 0, 3);
    v16h a = load_a_frag(Ab + k0, K, lane);
    acc0 = wmma_f16(a, load_b_frag(W + (size_t)(n0)*K      + k0, K, lane), acc0);
    acc1 = wmma_f16(a, load_b_frag(W + (size_t)(n0 + 16)*K + k0, K, lane), acc1);
    acc2 = wmma_f16(a, load_b_frag(W + (size_t)(n0 + 32)*K + k0, K, lane), acc2);
    acc3 = wmma_f16(a, load_b_frag(W + (size_t)(n0 + 48)*K + k0, K, lane), acc3);
  }

  const int nlo = lane & 15;
  const int mrow = (lane & 16) ? 8 : 0;
  if (mode == 0) {
    _Float16* C = (_Float16*)Cout + (size_t)b * strideC;
    for (int v = 0; v < 8; ++v) {
      size_t r = (size_t)(m0 + v + mrow) * N + n0 + nlo;
      C[r]      = (_Float16)acc0[v];
      C[r + 16] = (_Float16)acc1[v];
      C[r + 32] = (_Float16)acc2[v];
      C[r + 48] = (_Float16)acc3[v];
    }
  } else {
    float* C = (float*)Cout + (size_t)b * strideC;
    for (int v = 0; v < 8; ++v) {
      int m = m0 + v + mrow;
      C[(size_t)(n0 + nlo)      * M + m] = acc0[v];
      C[(size_t)(n0 + 16 + nlo) * M + m] = acc1[v];
      C[(size_t)(n0 + 32 + nlo) * M + m] = acc2[v];
      C[(size_t)(n0 + 48 + nlo) * M + m] = acc3[v];
    }
  }
}

// ---------------------------------------------------------------------------
// Flash-style attention with Euclidean-distance similarity.
// Q[b][i][512] f16 (head h cols h*64..h*64+63), KV[b][n][1024] f16 (K|V halves),
// q2/k2 [bh][1024] f32.  Out f16 [b][i][512].
// Block = 256 threads = 8 waves, each wave owns one 16-row query tile.
// j processed in tiles of 32; K tiles staged via global_load_async_to_lds_b128
// (ASYNCcnt) into double-buffered LDS so the DMA overlaps the previous tile's
// WMMAs + softmax.
// ---------------------------------------------------------------------------
__global__ __launch_bounds__(256) void attn_kernel(
    const _Float16* __restrict__ Q, const _Float16* __restrict__ KV,
    const float* __restrict__ q2, const float* __restrict__ k2,
    _Float16* __restrict__ Out) {
  __shared__ __align__(16) _Float16 Klds[2][32 * 64]; // K tile  [j][d], 2 bufs
  __shared__ __align__(16) _Float16 Vt[2][64 * 32];   // V tile transposed [d][j]
  __shared__ __align__(16) _Float16 Plds[8][16 * 32]; // per-wave P tile [m][j]
  __shared__ float k2l[2][32];

  const int tid = threadIdx.x;
  const int wave = tid >> 5, lane = tid & 31;
  const int bh = blockIdx.y, b = bh >> 3, h = bh & 7;
  const int i0 = blockIdx.x * 128 + wave * 16;
  const int nlo = lane & 15;
  const int mrow = (lane & 16) ? 8 : 0;
  const float scale = 0.125f;  // 64^-0.5

  // Cooperative-load geometry: each thread owns one 8-half (16B) chunk of the
  // 32x64 K tile and of the V tile.
  const int ldrow = tid >> 3;        // 0..31
  const int ldcol = (tid & 7) * 8;   // 0..56
  // LDS addresses for the async K-tile DMA (flat addr low 32 bits == LDS
  // offset per the CDNA5 flat-aperture mapping).
  unsigned ldsK[2];
  ldsK[0] = (unsigned)(uintptr_t)(&Klds[0][ldrow * 64 + ldcol]);
  ldsK[1] = (unsigned)(uintptr_t)(&Klds[1][ldrow * 64 + ldcol]);
  const _Float16* KVb = KV + ((size_t)(b * 1024 + ldrow)) * 1024 + h * 64 + ldcol;
  const float* k2b = k2 + (size_t)bh * 1024;

  // Stage tile at row offset j0 into buffer `buf`: async DMA for K, register
  // transpose for V, scalar slice for k2.
  auto stage = [&](int j0, int buf) {
    const _Float16* src = KVb + (size_t)j0 * 1024;
    asm volatile("global_load_async_to_lds_b128 %0, %1, off"
                 :: "v"(ldsK[buf]), "v"(src)
                 : "memory");
    v8h vv = *(const v8h*)(src + 512);
    _Float16* vt = &Vt[buf][0];
    for (int u = 0; u < 8; ++u) vt[(ldcol + u) * 32 + ldrow] = vv[u];
    if (tid < 32) k2l[buf][tid] = k2b[j0 + tid];
  };

  // Q A-fragments for the two d-chunks of 32 (resident for the whole j loop).
  const _Float16* Qbase = Q + ((size_t)(b * 1024 + i0)) * 512 + h * 64;
  v16h qa0 = load_a_frag(Qbase, 512, lane);
  v16h qa1 = load_a_frag(Qbase + 32, 512, lane);

  float qq[8];
  for (int v = 0; v < 8; ++v)
    qq[v] = q2[(size_t)bh * 1024 + i0 + v + mrow];

  float mi[8], li[8];
  v8f O0 = {}, O1 = {}, O2 = {}, O3 = {};
  for (int v = 0; v < 8; ++v) { mi[v] = -3.0e38f; li[v] = 0.f; }

  // Prologue: stage tile 0, wait, barrier.
  stage(0, 0);
  asm volatile("s_wait_asynccnt 0x0" ::: "memory");
  __syncthreads();

  for (int jt = 0; jt < 32; ++jt) {
    const int cur = jt & 1;
    // Kick off the next tile's DMA before computing on the current one.
    if (jt + 1 < 32) stage((jt + 1) * 32, cur ^ 1);

    const _Float16* Kc = &Klds[cur][0];
    const _Float16* Vc = &Vt[cur][0];

    // S = Q . K^T   (two 16x16 n-subtiles, two k-chunks of 32 each)
    v8f s0 = {}, s1 = {};
    s0 = wmma_f16(qa0, load_b_frag(Kc, 64, lane), s0);
    s0 = wmma_f16(qa1, load_b_frag(Kc + 32, 64, lane), s0);
    s1 = wmma_f16(qa0, load_b_frag(Kc + 16 * 64, 64, lane), s1);
    s1 = wmma_f16(qa1, load_b_frag(Kc + 16 * 64 + 32, 64, lane), s1);

    const float k2a = k2l[cur][nlo];
    const float k2c = k2l[cur][16 + nlo];
    _Float16* P = &Plds[wave][0];

    for (int v = 0; v < 8; ++v) {
      float d0 = fmaxf(qq[v] + k2a - 2.f * s0[v], 0.f);
      float d1 = fmaxf(qq[v] + k2c - 2.f * s1[v], 0.f);
      float sim0 = -scale * sqrtf(d0);
      float sim1 = -scale * sqrtf(d1);
      // Row (m = v + mrow) reduction: lanes 0-15 hold rows 0..7, 16-31 rows 8..15;
      // xor masks <=8 stay within each half (wave32).
      float mx = fmaxf(sim0, sim1);
      for (int off = 8; off > 0; off >>= 1)
        mx = fmaxf(mx, __shfl_xor(mx, off, 32));
      float mnew = fmaxf(mi[v], mx);
      float p0 = __expf(sim0 - mnew);
      float p1 = __expf(sim1 - mnew);
      float rs = p0 + p1;
      for (int off = 8; off > 0; off >>= 1)
        rs += __shfl_xor(rs, off, 32);
      float alpha = __expf(mi[v] - mnew);
      li[v] = li[v] * alpha + rs;
      mi[v] = mnew;
      O0[v] *= alpha; O1[v] *= alpha; O2[v] *= alpha; O3[v] *= alpha;
      P[(v + mrow) * 32 + nlo]      = (_Float16)p0;
      P[(v + mrow) * 32 + 16 + nlo] = (_Float16)p1;
    }

    // O += P . V   (A = P 16x32, B = V^T via transposed LDS tile; 4 d-subtiles)
    v16h pa = load_a_frag(P, 32, lane);
    O0 = wmma_f16(pa, load_b_frag(Vc + 0 * 16 * 32, 32, lane), O0);
    O1 = wmma_f16(pa, load_b_frag(Vc + 1 * 16 * 32, 32, lane), O1);
    O2 = wmma_f16(pa, load_b_frag(Vc + 2 * 16 * 32, 32, lane), O2);
    O3 = wmma_f16(pa, load_b_frag(Vc + 3 * 16 * 32, 32, lane), O3);

    // Next tile's DMA must have landed and every wave must be done with the
    // current buffers before they are recycled.
    asm volatile("s_wait_asynccnt 0x0" ::: "memory");
    __syncthreads();
  }

  _Float16* Ob = Out + ((size_t)(b * 1024 + i0)) * 512 + h * 64;
  for (int v = 0; v < 8; ++v) {
    float inv = 1.0f / li[v];
    size_t r = (size_t)(v + mrow) * 512 + nlo;
    Ob[r]      = (_Float16)(O0[v] * inv);
    Ob[r + 16] = (_Float16)(O1[v] * inv);
    Ob[r + 32] = (_Float16)(O2[v] * inv);
    Ob[r + 48] = (_Float16)(O3[v] * inv);
  }
}

// ---------------------------------------------------------------------------
// Host-side launcher
// ---------------------------------------------------------------------------
extern "C" void kernel_launch(void* const* d_in, const int* in_sizes, int n_in,
                              void* d_out, int out_size, void* d_ws, size_t ws_size,
                              hipStream_t stream) {
  const float* fmap       = (const float*)d_in[0];  // [8,512,32,32]
  const float* context    = (const float*)d_in[1];  // [8,1024,512]
  const float* gamma_fmap = (const float*)d_in[2];  // [512]
  const float* gamma_ctx  = (const float*)d_in[3];  // [512]
  const float* Wq         = (const float*)d_in[4];  // [512,512]
  const float* Wkv        = (const float*)d_in[5];  // [1024,512]
  const float* Wout       = (const float*)d_in[6];  // [512,512]
  float* out = (float*)d_out;                       // [8,512,32,32]

  char* ws = (char*)d_ws;
  auto alloc = [&](size_t bytes) -> char* {
    char* p = ws;
    ws += (bytes + 255) & ~(size_t)255;
    return p;
  };
  _Float16* fm_n   = (_Float16*)alloc(8ull * 1024 * 512 * 2);
  _Float16* ctx_n  = (_Float16*)alloc(8ull * 1024 * 512 * 2);
  _Float16* Wq16   = (_Float16*)alloc(512ull * 512 * 2);
  _Float16* Wkv16  = (_Float16*)alloc(1024ull * 512 * 2);
  _Float16* Wout16 = (_Float16*)alloc(512ull * 512 * 2);
  _Float16* Q16    = (_Float16*)alloc(8ull * 1024 * 512 * 2);
  _Float16* KV16   = (_Float16*)alloc(8ull * 1024 * 1024 * 2);
  _Float16* AT16   = (_Float16*)alloc(8ull * 1024 * 512 * 2);
  float* q2 = (float*)alloc(64ull * 1024 * 4);
  float* k2 = (float*)alloc(64ull * 1024 * 4);

  // Weight conversion (keep [N][K] layout: exactly the WMMA B-fragment layout).
  f32_to_f16_kernel<<<262144 / 256, 256, 0, stream>>>(Wq, Wq16, 262144);
  f32_to_f16_kernel<<<524288 / 256, 256, 0, stream>>>(Wkv, Wkv16, 524288);
  f32_to_f16_kernel<<<262144 / 256, 256, 0, stream>>>(Wout, Wout16, 262144);

  // Normalizations (fmap is transposed [b][c][xy] -> [b][xy][c] on the way out).
  norm_fmap_kernel<<<dim3(4, 8), 256, 0, stream>>>(fmap, gamma_fmap, fm_n);
  norm_rows_kernel<<<8192, 256, 0, stream>>>(context, gamma_ctx, ctx_n);

  // Projections.
  gemm_f16_kernel<<<dim3(8, 8, 8), 256, 0, stream>>>(
      fm_n, Wq16, Q16, 1024, 512, 512, 1024ull * 512, 1024ull * 512, 0);
  gemm_f16_kernel<<<dim3(16, 8, 8), 256, 0, stream>>>(
      ctx_n, Wkv16, KV16, 1024, 1024, 512, 1024ull * 512, 1024ull * 1024, 0);

  // Per-head squared norms for the cdist trick.
  rowsq_kernel<<<65536 / 256, 256, 0, stream>>>(Q16, q2, 512, 0);
  rowsq_kernel<<<65536 / 256, 256, 0, stream>>>(KV16, k2, 1024, 0);

  // Distance attention (flash-style, never materializes sim).
  attn_kernel<<<dim3(8, 64), 256, 0, stream>>>(Q16, KV16, q2, k2, AT16);

  // Output projection, stored transposed f32 into [b][dim][x][y].
  gemm_f16_kernel<<<dim3(8, 8, 8), 256, 0, stream>>>(
      AT16, Wout16, out, 1024, 512, 512, 1024ull * 512, 512ull * 1024, 1);
}
